// FastScaledDotProductAttention_39135742001244
// MI455X (gfx1250) — matrix operand order repaired and checked by reading, not compiled, not run on hardware
//
#include <hip/hip_runtime.h>

// ---------------------------------------------------------------------------
// Performer-style linear attention for MI455X (gfx1250, wave32, WMMA bf16).
//   B=4, H=16 -> 64 heads; C=E=D=128; N=M=4096.
// ---------------------------------------------------------------------------

#define DIMC   128
#define NSEQ   4096
#define HEADS  64           // B*H
#define NT     64           // n-tile for the kv pass
#define NCHUNK 8            // n-chunks per head (parallelism)
#define ITERS  ((NSEQ / NCHUNK) / NT)   // 8 tiles of 64 per workgroup
#define MT     64           // m-tile for the output pass
#define MCHUNKS (NSEQ / MT) // 64

// Padded LDS leading dims: lane-to-lane bank stride 4 (136) / 36 (72) dwords,
// both keep rows 16-byte aligned for b128 fragment loads.
#define LD128  136
#define LD64   72

typedef __bf16 bf16;
typedef __attribute__((ext_vector_type(16))) __bf16        v16bf;
typedef __attribute__((ext_vector_type(4)))  __bf16        v4bf;
typedef __attribute__((ext_vector_type(8)))  float         v8f;
typedef __attribute__((ext_vector_type(4)))  unsigned int  v4u;

union AFrag { v16bf v; unsigned int u[8]; };
union BFrag { v16bf v; v4u q[2]; };

static __device__ inline v8f vzero8() {
  v8f z = {0.f, 0.f, 0.f, 0.f, 0.f, 0.f, 0.f, 0.f};
  return z;
}

// A-matrix fragment (16x32 bf16, MxK) from row-major LDS [row][ld] (K contig).
// Lane<16 -> row M=lane, K = kbase+0..15; lane>=16 -> K = kbase+16..31.
static __device__ inline v16bf load_a_frag(const bf16* lds, int rowBase,
                                           int ld, int kbase, int lane) {
  AFrag f;
  const int r = rowBase + (lane & 15);
  const int half = (lane >> 4) & 1;
  const unsigned int* p =
      (const unsigned int*)(lds + r * ld + kbase + half * 16);
#pragma unroll
  for (int j = 0; j < 8; ++j) {
    const int koff = j * 2;  // relative pair start
    f.u[j] = p[koff >> 1];
  }
  return f.v;
}

// B-matrix fragment (32x16 bf16, KxN) from LDS laid out [n][ld], K contiguous.
// Lane<16 -> col N=lane, K=kbase+0..15; lane>=16 -> K=kbase+16..31.
static __device__ inline v16bf load_b_frag(const bf16* lds, int colBase,
                                           int ld, int kbase, int lane) {
  BFrag f;
  const int c = colBase + (lane & 15);
  const int half = (lane >> 4) & 1;
  const v4u* p = (const v4u*)(lds + c * ld + kbase + half * 16);
  f.q[0] = p[0];
  f.q[1] = p[1];
  return f.v;
}

static __device__ inline v8f wmma_bf16(v16bf a, v16bf b, v8f c) {
  return __builtin_amdgcn_wmma_f32_16x16x32_bf16(false, a, false, b,
                                                 (short)0, c, false, false);
}

// ---------------------------------------------------------------------------
// Kernel 0: zero the kv / rowsum scratch.
// ---------------------------------------------------------------------------
__global__ void fla_zero_kernel(float* __restrict__ p, int n) {
  const int i = blockIdx.x * 256 + threadIdx.x;
  if (i < n) p[i] = 0.f;
}

// ---------------------------------------------------------------------------
// Kernel 1: per head, per n-chunk:
//   phi_k = relu(P^T K),  kv += phi_k V^T,  rowsum += sum_n phi_k
// ---------------------------------------------------------------------------
__global__ __launch_bounds__(256) void
fla_kv_rowsum_kernel(const float* __restrict__ Kg, const float* __restrict__ Vg,
                     const float* __restrict__ Pg, float* __restrict__ kv_ws,
                     float* __restrict__ rs_ws) {
  __shared__ alignas(16) bf16 ptl[DIMC * LD128];  // P^T      [e][c]  34.0 KB
  __shared__ alignas(16) bf16 ktl[NT * LD128];    // K^T tile [n][c]  17.0 KB
  __shared__ alignas(16) bf16 vtl[DIMC * LD64];   // V tile   [d][n]  18.0 KB
  __shared__ alignas(16) bf16 phil[DIMC * LD64];  // phi tile [e][n]  18.0 KB

  const int head = blockIdx.y;
  const int chunk = blockIdx.x;
  const int tid = threadIdx.x;
  const int lane = tid & 31;
  const int wave = tid >> 5;
  const int half = (lane >> 4) & 1;

  const float* __restrict__ Kh = Kg + (size_t)head * DIMC * NSEQ;
  const float* __restrict__ Vh = Vg + (size_t)head * DIMC * NSEQ;

  // Stage P^T once: b128 reads along e, bf16 transposed writes.
  for (int i = tid; i < (DIMC * DIMC) / 4; i += 256) {
    const int e4 = i & 31;
    const int c = i >> 5;
    const float4 pq = *(const float4*)(Pg + c * DIMC + e4 * 4);
    ptl[(e4 * 4 + 0) * LD128 + c] = (bf16)pq.x;
    ptl[(e4 * 4 + 1) * LD128 + c] = (bf16)pq.y;
    ptl[(e4 * 4 + 2) * LD128 + c] = (bf16)pq.z;
    ptl[(e4 * 4 + 3) * LD128 + c] = (bf16)pq.w;
  }

  // Per-wave kv accumulators: wave owns rows e in [16*wave, 16*wave+16),
  // all 8 d-tiles -> full 128x128 across 8 waves.
  v8f kvacc[8];
#pragma unroll
  for (int t = 0; t < 8; ++t) kvacc[t] = vzero8();
  float rs = 0.f;
  const int e_rs = tid & (DIMC - 1);
  const int part = tid >> 7;  // 0/1: which 32 columns of the tile to sum

  __syncthreads();

  const int nbase = chunk * (NSEQ / NCHUNK);
  for (int it = 0; it < ITERS; ++it) {
    const int n0 = nbase + it * NT;

    // Stage K^T [n][c] (transposed) and V [d][n] via float4 (b128) reads.
    for (int i = tid; i < (DIMC * NT) / 4; i += 256) {
      const int n4 = i & (NT / 4 - 1);  // 0..15
      const int c = i / (NT / 4);       // 0..127 (c for K, d for V)
      const float4 kq = *(const float4*)(Kh + (size_t)c * NSEQ + n0 + n4 * 4);
      const float4 vq = *(const float4*)(Vh + (size_t)c * NSEQ + n0 + n4 * 4);
      ktl[(n4 * 4 + 0) * LD128 + c] = (bf16)kq.x;
      ktl[(n4 * 4 + 1) * LD128 + c] = (bf16)kq.y;
      ktl[(n4 * 4 + 2) * LD128 + c] = (bf16)kq.z;
      ktl[(n4 * 4 + 3) * LD128 + c] = (bf16)kq.w;
      v4bf vp = {(bf16)vq.x, (bf16)vq.y, (bf16)vq.z, (bf16)vq.w};
      *(v4bf*)(vtl + c * LD64 + n4 * 4) = vp;  // 8-byte ds_store
    }
    if (it + 1 < ITERS) {
      __builtin_prefetch(&Kh[(size_t)(tid & (DIMC - 1)) * NSEQ + n0 + NT], 0, 0);
      __builtin_prefetch(&Vh[(size_t)(tid & (DIMC - 1)) * NSEQ + n0 + NT], 0, 0);
    }
    __syncthreads();

    // GEMM1: phi = P^T(128x128) @ Ktile(128x64); wave -> rows 16*wave.
    v8f ph[4];
#pragma unroll
    for (int nt = 0; nt < 4; ++nt) ph[nt] = vzero8();
#pragma unroll
    for (int kk = 0; kk < 4; ++kk) {
      const v16bf a = load_a_frag(ptl, wave * 16, LD128, kk * 32, lane);
#pragma unroll
      for (int nt = 0; nt < 4; ++nt) {
        const v16bf b = load_b_frag(ktl, nt * 16, LD128, kk * 32, lane);
        ph[nt] = wmma_bf16(a, b, ph[nt]);
      }
    }

    // ReLU and spill phi -> LDS as [e][n] (A-operand layout for GEMM2).
#pragma unroll
    for (int nt = 0; nt < 4; ++nt) {
      const int n = nt * 16 + (lane & 15);
#pragma unroll
      for (int r = 0; r < 8; ++r) {
        float x = ph[nt][r];
        x = x > 0.f ? x : 0.f;
        const int e = wave * 16 + half * 8 + r;
        phil[e * LD64 + n] = (bf16)x;
      }
    }
    __syncthreads();

    // GEMM2: kv(128x128) += phi(128x64) @ V^T(64x128).
#pragma unroll
    for (int kk = 0; kk < 2; ++kk) {
      const v16bf a = load_a_frag(phil, wave * 16, LD64, kk * 32, lane);
#pragma unroll
      for (int dt = 0; dt < 8; ++dt) {
        const v16bf b = load_b_frag(vtl, dt * 16, LD64, kk * 32, lane);
        kvacc[dt] = wmma_bf16(a, b, kvacc[dt]);
      }
    }

    // rowsum partial: thread (e, part) sums 32 columns of phi row e.
    {
      const bf16* pr = phil + e_rs * LD64 + part * 32;
      float s = 0.f;
#pragma unroll
      for (int n = 0; n < 32; ++n) s += (float)pr[n];
      rs += s;
    }
    __syncthreads();
  }

  // Flush accumulators (fp32 atomics into scratch).
  float* __restrict__ kvh = kv_ws + (size_t)head * DIMC * DIMC;
  const int dcol = lane & 15;
#pragma unroll
  for (int dt = 0; dt < 8; ++dt) {
#pragma unroll
    for (int r = 0; r < 8; ++r) {
      const int e = wave * 16 + half * 8 + r;
      const int d = dt * 16 + dcol;
      atomicAdd(&kvh[e * DIMC + d], kvacc[dt][r]);
    }
  }
  atomicAdd(&rs_ws[head * DIMC + e_rs], rs);
}

// ---------------------------------------------------------------------------
// Kernel 2: per head, per m-tile:
//   phi_q = relu(P^T Q/sqrt(128)); denom = phi_q^T rowsum;
//   out   = (kv^T @ phi_q) * (1/(denom+1e-8))
// LDS phase-aliased: P^T -> kv^T -> f32 out tile;  Qtile -> phi_q.
// ---------------------------------------------------------------------------
__global__ __launch_bounds__(256) void
fla_out_kernel(const float* __restrict__ Qg, const float* __restrict__ Pg,
               const float* __restrict__ kv_ws, const float* __restrict__ rs_ws,
               float* __restrict__ Og) {
  __shared__ alignas(16) bf16 smem[DIMC * LD128 + MT * LD128];  // ~51 KB
  __shared__ float rsl[DIMC];
  __shared__ float scl[MT];

  bf16* regA = smem;                // P^T [e][c] -> kv^T [d][e] -> f32 out
  bf16* regB = smem + DIMC * LD128; // q [m][c]   -> phiq [m][e]

  const int head = blockIdx.y;
  const int m0 = blockIdx.x * MT;
  const int tid = threadIdx.x;
  const int lane = tid & 31;
  const int wave = tid >> 5;
  const int half = (lane >> 4) & 1;

  const float* __restrict__ Qh = Qg + (size_t)head * DIMC * NSEQ;
  const float* __restrict__ kvh = kv_ws + (size_t)head * DIMC * DIMC;
  const float qscale = 0.08838834764831845f;  // 1/sqrt(128)

  // Stage P^T (b128 reads along e) and scaled Q tile (b128 reads along m).
  for (int i = tid; i < (DIMC * DIMC) / 4; i += 256) {
    const int e4 = i & 31;
    const int c = i >> 5;
    const float4 pq = *(const float4*)(Pg + c * DIMC + e4 * 4);
    regA[(e4 * 4 + 0) * LD128 + c] = (bf16)pq.x;
    regA[(e4 * 4 + 1) * LD128 + c] = (bf16)pq.y;
    regA[(e4 * 4 + 2) * LD128 + c] = (bf16)pq.z;
    regA[(e4 * 4 + 3) * LD128 + c] = (bf16)pq.w;
  }
  for (int i = tid; i < (DIMC * MT) / 4; i += 256) {
    const int m4 = i & (MT / 4 - 1);
    const int c = i / (MT / 4);
    const float4 qq = *(const float4*)(Qh + (size_t)c * NSEQ + m0 + m4 * 4);
    regB[(m4 * 4 + 0) * LD128 + c] = (bf16)(qq.x * qscale);
    regB[(m4 * 4 + 1) * LD128 + c] = (bf16)(qq.y * qscale);
    regB[(m4 * 4 + 2) * LD128 + c] = (bf16)(qq.z * qscale);
    regB[(m4 * 4 + 3) * LD128 + c] = (bf16)(qq.w * qscale);
  }
  if (tid < DIMC) rsl[tid] = rs_ws[head * DIMC + tid];
  __syncthreads();

  // GEMM: phiq = P^T(128x128) @ q(128x64); wave -> rows e in [16w,16w+16).
  v8f ph[4];
#pragma unroll
  for (int mt = 0; mt < 4; ++mt) ph[mt] = vzero8();
#pragma unroll
  for (int kk = 0; kk < 4; ++kk) {
    const v16bf a = load_a_frag(regA, wave * 16, LD128, kk * 32, lane);
#pragma unroll
    for (int mt = 0; mt < 4; ++mt) {
      const v16bf b = load_b_frag(regB, mt * 16, LD128, kk * 32, lane);
      ph[mt] = wmma_bf16(a, b, ph[mt]);
    }
  }
  __syncthreads();  // all waves done with regA/regB contents

  // Phase 2 staging: phiq (relu) -> regB as [m][e]; kv^T -> regA as [d][e].
#pragma unroll
  for (int mt = 0; mt < 4; ++mt) {
    const int m = mt * 16 + (lane & 15);
#pragma unroll
    for (int r = 0; r < 8; ++r) {
      float x = ph[mt][r];
      x = x > 0.f ? x : 0.f;
      const int e = wave * 16 + half * 8 + r;
      regB[m * LD128 + e] = (bf16)x;
    }
  }
  for (int i = tid; i < (DIMC * DIMC) / 4; i += 256) {
    const int d4 = i & 31;
    const int e = i >> 5;
    const float4 kq = *(const float4*)(kvh + e * DIMC + d4 * 4);
    regA[(d4 * 4 + 0) * LD128 + e] = (bf16)kq.x;
    regA[(d4 * 4 + 1) * LD128 + e] = (bf16)kq.y;
    regA[(d4 * 4 + 2) * LD128 + e] = (bf16)kq.z;
    regA[(d4 * 4 + 3) * LD128 + e] = (bf16)kq.w;
  }
  __syncthreads();

  // denom[m] = sum_e phiq[m][e] * rowsum[e];  scale = 1/(denom + 1e-8)
  if (tid < MT) {
    const bf16* pm = regB + tid * LD128;
    float s = 0.f;
#pragma unroll 8
    for (int e = 0; e < DIMC; ++e) s += (float)pm[e] * rsl[e];
    scl[tid] = 1.f / (s + 1e-8f);
  }
  __syncthreads();

  // GEMM: out = kv^T(128x128) @ phiq(128x64); wave -> rows d in [16w,16w+16).
  v8f oc[4];
#pragma unroll
  for (int mt = 0; mt < 4; ++mt) oc[mt] = vzero8();
#pragma unroll
  for (int kk = 0; kk < 4; ++kk) {
    const v16bf a = load_a_frag(regA, wave * 16, LD128, kk * 32, lane);
#pragma unroll
    for (int mt = 0; mt < 4; ++mt) {
      const v16bf b = load_b_frag(regB, mt * 16, LD128, kk * 32, lane);
      oc[mt] = wmma_bf16(a, b, oc[mt]);
    }
  }
  __syncthreads();  // everyone done reading regA before f32 alias write

  // Scale and stage the f32 out tile in LDS, then store coalesced b128 rows.
  float* outl = (float*)smem;  // 128*64*4 = 32 KB, fits the regA region
#pragma unroll
  for (int mt = 0; mt < 4; ++mt) {
    const int mloc = mt * 16 + (lane & 15);
    const float sc = scl[mloc];
#pragma unroll
    for (int r = 0; r < 8; ++r) {
      const int d = wave * 16 + half * 8 + r;
      outl[d * MT + mloc] = oc[mt][r] * sc;
    }
  }
  __syncthreads();

  float* __restrict__ Oh = Og + (size_t)head * DIMC * NSEQ;
  for (int i = tid; i < (DIMC * MT) / 4; i += 256) {
    const int m4 = i & (MT / 4 - 1);
    const int d = i / (MT / 4);
    const float4 o = *(const float4*)(outl + d * MT + m4 * 4);
    *(float4*)(Oh + (size_t)d * NSEQ + m0 + m4 * 4) = o;
  }
}

// ---------------------------------------------------------------------------
extern "C" void kernel_launch(void* const* d_in, const int* in_sizes, int n_in,
                              void* d_out, int out_size, void* d_ws,
                              size_t ws_size, hipStream_t stream) {
  (void)in_sizes; (void)n_in; (void)out_size; (void)ws_size;

  const float* q = (const float*)d_in[0];     // [B,H,128,4096]
  const float* k = (const float*)d_in[1];     // [B,H,128,4096]
  const float* v = (const float*)d_in[2];     // [B,H,128,4096]
  const float* p = (const float*)d_in[3];     // [128,128]
  float* out = (float*)d_out;                 // [B,H,128,4096]

  float* kv_ws = (float*)d_ws;                           // 64*128*128 f32
  float* rs_ws = kv_ws + (size_t)HEADS * DIMC * DIMC;    // 64*128 f32

  const int nz = HEADS * DIMC * DIMC + HEADS * DIMC;
  fla_zero_kernel<<<(nz + 255) / 256, 256, 0, stream>>>(kv_ws, nz);

  fla_kv_rowsum_kernel<<<dim3(NCHUNK, HEADS), dim3(256), 0, stream>>>(
      k, v, p, kv_ws, rs_ws);

  fla_out_kernel<<<dim3(MCHUNKS, HEADS), dim3(256), 0, stream>>>(
      q, p, kv_ws, rs_ws, out);
}
